// RSSM_32220844654742
// MI455X (gfx1250) — matrix operand order repaired and checked by reading, not compiled, not run on hardware
//
#include <hip/hip_runtime.h>
#include <hip/hip_bf16.h>

#define B_   32
#define T_   32
#define VIS  1024
#define FF   512
#define LAT  512
#define ACT  8
#define ACAT 544   // (LAT + ACT) zero-padded up to a multiple of 32

typedef __bf16 bf16;
typedef __attribute__((ext_vector_type(8)))  __bf16 v8bf;
typedef __attribute__((ext_vector_type(16))) __bf16 v16bf;
typedef __attribute__((ext_vector_type(8)))  float  v8f;

__device__ __forceinline__ v8f zero8() {
    v8f z = {0.f,0.f,0.f,0.f,0.f,0.f,0.f,0.f};
    return z;
}
__device__ __forceinline__ float sigmoidf_(float x) {
    return 1.0f / (1.0f + expf(-x));
}
__device__ __forceinline__ float softplusf_(float x) {
    return fmaxf(x, 0.0f) + log1pf(expf(-fabsf(x)));   // jax.nn.softplus
}

// bf16 WMMA accumulate:  acc += A(16x32) * B(32x16)
__device__ __forceinline__ v8f bmma(v8f acc, v16bf a, v16bf b) {
    return __builtin_amdgcn_wmma_f32_16x16x32_bf16(
        false, a, false, b, (short)0, acc, false, false);
}
// bf16x3 fp32-accurate product: hi*hi + hi*lo + lo*hi
__device__ __forceinline__ v8f bmma3(v8f acc, v16bf ah, v16bf al, v16bf bh, v16bf bl) {
    acc = bmma(acc, ah, bh);
    acc = bmma(acc, ah, bl);
    acc = bmma(acc, al, bh);
    return acc;
}

// A-fragment (16-bit A 16x32 layout): runs [kc+8h,+8) and [kc+16+8h,+8)
__device__ __forceinline__ v16bf afrag(const bf16* __restrict__ row, int kc, int half) {
    v8bf a0 = *(const v8bf*)(row + kc + 8 * half);
    v8bf a1 = *(const v8bf*)(row + kc + 16 + 8 * half);
    return __builtin_shufflevector(a0, a1, 0,1,2,3,4,5,6,7,8,9,10,11,12,13,14,15);
}
// B-fragment (16-bit B 32x16 layout): runs [kc+8h,+8) and [kc+16+8h,+8) of W row n
__device__ __forceinline__ v16bf bfrag(const bf16* __restrict__ wrow, int kc, int half) {
    v8bf b0 = *(const v8bf*)(wrow + kc + 8 * half);
    v8bf b1 = *(const v8bf*)(wrow + kc + 16 + 8 * half);
    return __builtin_shufflevector(b0, b1, 0,1,2,3,4,5,6,7,8,9,10,11,12,13,14,15);
}
__device__ __forceinline__ void split1(float v, bf16* h, bf16* l) {
    bf16 hi = (bf16)v;
    *h = hi;
    *l = (bf16)(v - (float)hi);
}

// ---------------------------------------------------------------------------
// mean pool over trailing 48 elements, emit split-bf16 directly
// ---------------------------------------------------------------------------
__global__ void pool_split(const float* __restrict__ img,
                           bf16* __restrict__ ph, bf16* __restrict__ pl, int n) {
    int i = blockIdx.x * blockDim.x + threadIdx.x;
    if (i >= n) return;
    const float4* p = (const float4*)(img + (size_t)i * 48);
    float s = 0.f;
#pragma unroll
    for (int j = 0; j < 12; ++j) {
        float4 v = p[j];
        s += v.x + v.y + v.z + v.w;
    }
    split1(s * (1.0f / 48.0f), ph + i, pl + i);
}

// ---------------------------------------------------------------------------
// split fp32 matrix (R x Ksrc, row stride lds) into hi/lo bf16, K-padded to ldd
// ---------------------------------------------------------------------------
__global__ void split_pad(const float* __restrict__ src, int R, int Ksrc, int lds,
                          int ldd, bf16* __restrict__ hi, bf16* __restrict__ lo) {
    int i = blockIdx.x * blockDim.x + threadIdx.x;
    if (i >= R * ldd) return;
    int rr = i / ldd, c = i - rr * ldd;
    float v = (c < Ksrc) ? src[(size_t)rr * lds + c] : 0.f;
    split1(v, hi + i, lo + i);
}

// ---------------------------------------------------------------------------
// generic C = act( [A1 | A2] @ W^T + bias ) via bf16x3 WMMA.
// K1, K2 multiples of 32. Optional fp32 and/or split-bf16 outputs.
// ---------------------------------------------------------------------------
__global__ void gemm_bf3(const bf16* __restrict__ A1h, const bf16* __restrict__ A1l,
                         int s1, int K1,
                         const bf16* __restrict__ A2h, const bf16* __restrict__ A2l,
                         int s2, int K2,
                         const bf16* __restrict__ Wh, const bf16* __restrict__ Wl,
                         int ldw, const float* __restrict__ bias,
                         float* __restrict__ Cf, int ldcf,
                         bf16* __restrict__ Chh, bf16* __restrict__ Cll, int ldcb,
                         int Mtiles, int Ntiles, int act) {
    int wave = (blockIdx.x * blockDim.x + threadIdx.x) >> 5;
    int lane = threadIdx.x & 31;
    if (wave >= Mtiles * Ntiles) return;
    int mt = wave % Mtiles, nt = wave / Mtiles;
    int m0 = mt * 16, n0 = nt * 16;
    int r = lane & 15, half = lane >> 4;

    const bf16* a1h = A1h + (size_t)(m0 + r) * s1;
    const bf16* a1l = A1l + (size_t)(m0 + r) * s1;
    const bf16* wh  = Wh + (size_t)(n0 + r) * ldw;
    const bf16* wl  = Wl + (size_t)(n0 + r) * ldw;

    v8f acc = zero8();
    for (int kc = 0; kc < K1; kc += 32)
        acc = bmma3(acc, afrag(a1h, kc, half), afrag(a1l, kc, half),
                    bfrag(wh, kc, half), bfrag(wl, kc, half));
    if (K2 > 0) {
        const bf16* a2h = A2h + (size_t)(m0 + r) * s2;
        const bf16* a2l = A2l + (size_t)(m0 + r) * s2;
        for (int kc = 0; kc < K2; kc += 32)
            acc = bmma3(acc, afrag(a2h, kc, half), afrag(a2l, kc, half),
                        bfrag(wh, K1 + kc, half), bfrag(wl, K1 + kc, half));
    }

    int n = n0 + r;
    float bv = bias[n];
#pragma unroll
    for (int v = 0; v < 8; ++v) {
        int m = m0 + v + 8 * half;
        float x = acc[v] + bv;
        if (act) x = fmaxf(x, 0.f);
        if (Cf) Cf[(size_t)m * ldcf + n] = x;
        if (Chh) split1(x, Chh + (size_t)m * ldcb + n, Cll + (size_t)m * ldcb + n);
    }
}

// ---------------------------------------------------------------------------
// fused GRU step, bf16x3: gx = gi@Wih^T+bih, gh = h@Whh^T+bhh, gates, h_new.
// ---------------------------------------------------------------------------
__global__ void gru_bf3(const bf16* __restrict__ gih, const bf16* __restrict__ gil,
                        const bf16* __restrict__ hsh, const bf16* __restrict__ hsl,
                        const float* __restrict__ hf,
                        const bf16* __restrict__ Wih_h, const bf16* __restrict__ Wih_l,
                        const bf16* __restrict__ Whh_h, const bf16* __restrict__ Whh_l,
                        const float* __restrict__ bih, const float* __restrict__ bhh,
                        float* __restrict__ hf_new,
                        bf16* __restrict__ hh_new, bf16* __restrict__ hl_new) {
    int wave = (blockIdx.x * blockDim.x + threadIdx.x) >> 5;
    int lane = threadIdx.x & 31;
    if (wave >= 2 * (FF / 16)) return;
    int mt = wave & 1, nt = wave >> 1;
    int m0 = mt * 16, n0 = nt * 16;
    int r = lane & 15, half = lane >> 4;

    const bf16* gah = gih + (size_t)(m0 + r) * FF;
    const bf16* gal = gil + (size_t)(m0 + r) * FF;
    const bf16* hah = hsh + (size_t)(m0 + r) * FF;
    const bf16* hal = hsl + (size_t)(m0 + r) * FF;
    const bf16* wr[3], *wrl[3], *ur[3], *url[3];
#pragma unroll
    for (int j = 0; j < 3; ++j) {
        wr[j]  = Wih_h + (size_t)(n0 + r + j * FF) * FF;
        wrl[j] = Wih_l + (size_t)(n0 + r + j * FF) * FF;
        ur[j]  = Whh_h + (size_t)(n0 + r + j * FF) * FF;
        url[j] = Whh_l + (size_t)(n0 + r + j * FF) * FF;
    }

    v8f ax0 = zero8(), ax1 = zero8(), ax2 = zero8();
    v8f ah0 = zero8(), ah1 = zero8(), ah2 = zero8();
    for (int kc = 0; kc < FF; kc += 32) {
        v16bf g_h = afrag(gah, kc, half), g_l = afrag(gal, kc, half);
        v16bf h_h = afrag(hah, kc, half), h_l = afrag(hal, kc, half);
        ax0 = bmma3(ax0, g_h, g_l, bfrag(wr[0], kc, half), bfrag(wrl[0], kc, half));
        ax1 = bmma3(ax1, g_h, g_l, bfrag(wr[1], kc, half), bfrag(wrl[1], kc, half));
        ax2 = bmma3(ax2, g_h, g_l, bfrag(wr[2], kc, half), bfrag(wrl[2], kc, half));
        ah0 = bmma3(ah0, h_h, h_l, bfrag(ur[0], kc, half), bfrag(url[0], kc, half));
        ah1 = bmma3(ah1, h_h, h_l, bfrag(ur[1], kc, half), bfrag(url[1], kc, half));
        ah2 = bmma3(ah2, h_h, h_l, bfrag(ur[2], kc, half), bfrag(url[2], kc, half));
    }

    int n = n0 + r;
    float bxr = bih[n], bxz = bih[n + FF], bxn = bih[n + 2 * FF];
    float bhr = bhh[n], bhz = bhh[n + FF], bhn = bhh[n + 2 * FF];
#pragma unroll
    for (int v = 0; v < 8; ++v) {
        int m = m0 + v + 8 * half;
        float rg = sigmoidf_((ax0[v] + bxr) + (ah0[v] + bhr));
        float zg = sigmoidf_((ax1[v] + bxz) + (ah1[v] + bhz));
        float ng = tanhf((ax2[v] + bxn) + rg * (ah2[v] + bhn));
        float hv = hf[(size_t)m * FF + n];
        float hn2 = (1.f - zg) * ng + zg * hv;
        hf_new[(size_t)m * FF + n] = hn2;
        split1(hn2, hh_new + (size_t)m * FF + n, hl_new + (size_t)m * FF + n);
    }
}

// ---------------------------------------------------------------------------
// fused sampling head: s = (mid@Wm^T+bm) + (softplus(mid@Ws^T+bs)+0.01)*eps
// ---------------------------------------------------------------------------
__global__ void sample_bf3(const bf16* __restrict__ mh, const bf16* __restrict__ ml,
                           const bf16* __restrict__ Wmh, const bf16* __restrict__ Wml,
                           const float* __restrict__ bm,
                           const bf16* __restrict__ Wsh, const bf16* __restrict__ Wsl,
                           const float* __restrict__ bs,
                           const float* __restrict__ eps, int eps_rs,
                           float* __restrict__ out, int out_rs) {
    int wave = (blockIdx.x * blockDim.x + threadIdx.x) >> 5;
    int lane = threadIdx.x & 31;
    if (wave >= 2 * (LAT / 16)) return;
    int mt = wave & 1, nt = wave >> 1;
    int m0 = mt * 16, n0 = nt * 16;
    int r = lane & 15, half = lane >> 4;

    const bf16* ah = mh + (size_t)(m0 + r) * FF;
    const bf16* al = ml + (size_t)(m0 + r) * FF;
    const bf16* wm_h = Wmh + (size_t)(n0 + r) * FF;
    const bf16* wm_l = Wml + (size_t)(n0 + r) * FF;
    const bf16* ws_h = Wsh + (size_t)(n0 + r) * FF;
    const bf16* ws_l = Wsl + (size_t)(n0 + r) * FF;

    v8f am = zero8(), as = zero8();
    for (int kc = 0; kc < FF; kc += 32) {
        v16bf a_h = afrag(ah, kc, half), a_l = afrag(al, kc, half);
        am = bmma3(am, a_h, a_l, bfrag(wm_h, kc, half), bfrag(wm_l, kc, half));
        as = bmma3(as, a_h, a_l, bfrag(ws_h, kc, half), bfrag(ws_l, kc, half));
    }

    int n = n0 + r;
    float bmv = bm[n], bsv = bs[n];
#pragma unroll
    for (int v = 0; v < 8; ++v) {
        int m = m0 + v + 8 * half;
        float mean = am[v] + bmv;
        float sp   = softplusf_(as[v] + bsv) + 0.01f;
        out[(size_t)m * out_rs + n] = mean + sp * eps[(size_t)m * eps_rs + n];
    }
}

// ---------------------------------------------------------------------------
// build padded concat [s_prev | a_t | 0] as split bf16 (32 x 544)
// ---------------------------------------------------------------------------
__global__ void build_acat(const float* __restrict__ s, int s_rs,
                           const float* __restrict__ act_t, int act_rs,
                           bf16* __restrict__ a_hi, bf16* __restrict__ a_lo) {
    int i = blockIdx.x * blockDim.x + threadIdx.x;
    if (i >= B_ * ACAT) return;
    int b = i / ACAT, c = i - b * ACAT;
    float v = 0.f;
    if (c < LAT)            v = s[(size_t)b * s_rs + c];
    else if (c < LAT + ACT) v = act_t[(size_t)b * act_rs + (c - LAT)];
    split1(v, a_hi + i, a_lo + i);
}

// ---------------------------------------------------------------------------
// h0 = broadcast(init_hidden): fp32 + split bf16
// ---------------------------------------------------------------------------
__global__ void h0_init(const float* __restrict__ init_h, float* __restrict__ hf,
                        bf16* __restrict__ hh, bf16* __restrict__ hl, int n) {
    int i = blockIdx.x * blockDim.x + threadIdx.x;
    if (i >= n) return;
    float v = init_h[i & (FF - 1)];
    hf[i] = v;
    split1(v, hh + i, hl + i);
}

extern "C" void kernel_launch(void* const* d_in, const int* in_sizes, int n_in,
                              void* d_out, int out_size, void* d_ws, size_t ws_size,
                              hipStream_t stream) {
    const float* img     = (const float*)d_in[0];
    const float* actions = (const float*)d_in[1];
    const float* eps0    = (const float*)d_in[2];
    const float* eps_seq = (const float*)d_in[3];
    const float* init_h  = (const float*)d_in[4];
    const float* W_obs   = (const float*)d_in[5];
    const float* b_obs   = (const float*)d_in[6];
    const float* W_post1 = (const float*)d_in[7];
    const float* b_post1 = (const float*)d_in[8];
    const float* W_pm    = (const float*)d_in[9];
    const float* b_pm    = (const float*)d_in[10];
    const float* W_ps    = (const float*)d_in[11];
    const float* b_ps    = (const float*)d_in[12];
    const float* W_gi    = (const float*)d_in[13];
    const float* b_gi    = (const float*)d_in[14];
    const float* Wih     = (const float*)d_in[15];
    const float* Whh     = (const float*)d_in[16];
    const float* bih     = (const float*)d_in[17];
    const float* bhh     = (const float*)d_in[18];
    const float* W_pr1   = (const float*)d_in[19];
    const float* b_pr1   = (const float*)d_in[20];
    const float* W_prm   = (const float*)d_in[21];
    const float* b_prm   = (const float*)d_in[22];
    const float* W_prs   = (const float*)d_in[23];
    const float* b_prs   = (const float*)d_in[24];

    // ---- bump allocator over d_ws (64B aligned partitions)
    char* wp = (char*)d_ws;
    auto alloc = [&](size_t bytes) -> char* {
        char* ret = wp;
        wp += (bytes + 63) & ~(size_t)63;
        return ret;
    };
    auto allocB = [&](size_t elems) { return (bf16*)alloc(elems * sizeof(bf16)); };
    auto allocF = [&](size_t elems) { return (float*)alloc(elems * sizeof(float)); };

    // fp32 state
    float* hA_f = allocF((size_t)B_ * FF);
    float* hB_f = allocF((size_t)B_ * FF);
    // split-bf16 activations
    bf16 *pool_h = allocB((size_t)B_*T_*VIS), *pool_l = allocB((size_t)B_*T_*VIS);
    bf16 *obs_h  = allocB((size_t)B_*T_*FF),  *obs_l  = allocB((size_t)B_*T_*FF);
    bf16 *gi_h   = allocB((size_t)B_*FF),     *gi_l   = allocB((size_t)B_*FF);
    bf16 *mid_h  = allocB((size_t)B_*FF),     *mid_l  = allocB((size_t)B_*FF);
    bf16 *hA_h   = allocB((size_t)B_*FF),     *hA_l   = allocB((size_t)B_*FF);
    bf16 *hB_h   = allocB((size_t)B_*FF),     *hB_l   = allocB((size_t)B_*FF);
    bf16 *ac_h   = allocB((size_t)B_*ACAT),   *ac_l   = allocB((size_t)B_*ACAT);
    // split-bf16 weights (done once per launch; L2-resident for the scan)
    bf16 *wobs_h = allocB((size_t)FF*VIS),    *wobs_l = allocB((size_t)FF*VIS);
    bf16 *wp1_h  = allocB((size_t)FF*2*FF),   *wp1_l  = allocB((size_t)FF*2*FF);
    bf16 *wpm_h  = allocB((size_t)LAT*FF),    *wpm_l  = allocB((size_t)LAT*FF);
    bf16 *wps_h  = allocB((size_t)LAT*FF),    *wps_l  = allocB((size_t)LAT*FF);
    bf16 *wgi_h  = allocB((size_t)FF*ACAT),   *wgi_l  = allocB((size_t)FF*ACAT);
    bf16 *wih_h  = allocB((size_t)3*FF*FF),   *wih_l  = allocB((size_t)3*FF*FF);
    bf16 *whh_h  = allocB((size_t)3*FF*FF),   *whh_l  = allocB((size_t)3*FF*FF);
    bf16 *wpr1_h = allocB((size_t)FF*FF),     *wpr1_l = allocB((size_t)FF*FF);
    bf16 *wprm_h = allocB((size_t)LAT*FF),    *wprm_l = allocB((size_t)LAT*FF);
    bf16 *wprs_h = allocB((size_t)LAT*FF),    *wprs_l = allocB((size_t)LAT*FF);

    float* out = (float*)d_out;
    const int out_rs = (T_ + 1) * LAT;

    // ---- 1) pooling (HBM-bound part: 201 MB of img_feat)
    {
        int n = B_ * T_ * VIS;
        pool_split<<<(n + 255) / 256, 256, 0, stream>>>(img, pool_h, pool_l, n);
    }
    // ---- 2) split all weights once
    auto split_launch = [&](const float* src, int R, int Ksrc, int ldd,
                            bf16* hi, bf16* lo) {
        int n = R * ldd;
        split_pad<<<(n + 255) / 256, 256, 0, stream>>>(src, R, Ksrc, Ksrc, ldd, hi, lo);
    };
    split_launch(W_obs,   FF,     VIS,       VIS,  wobs_h, wobs_l);
    split_launch(W_post1, FF,     2 * FF,    2*FF, wp1_h,  wp1_l);
    split_launch(W_pm,    LAT,    FF,        FF,   wpm_h,  wpm_l);
    split_launch(W_ps,    LAT,    FF,        FF,   wps_h,  wps_l);
    split_launch(W_gi,    FF,     LAT + ACT, ACAT, wgi_h,  wgi_l);   // pad 520->544
    split_launch(Wih,     3 * FF, FF,        FF,   wih_h,  wih_l);
    split_launch(Whh,     3 * FF, FF,        FF,   whh_h,  whh_l);
    split_launch(W_pr1,   FF,     FF,        FF,   wpr1_h, wpr1_l);
    split_launch(W_prm,   LAT,    FF,        FF,   wprm_h, wprm_l);
    split_launch(W_prs,   LAT,    FF,        FF,   wprs_h, wprs_l);

    // ---- 3) obs_enc = pooled @ W_obs^T + b_obs  (1024 x 1024 x 512), split out
    {
        int waves = (B_ * T_ / 16) * (FF / 16);
        gemm_bf3<<<(waves * 32 + 127) / 128, 128, 0, stream>>>(
            pool_h, pool_l, VIS, VIS, nullptr, nullptr, 0, 0,
            wobs_h, wobs_l, VIS, b_obs,
            nullptr, 0, obs_h, obs_l, FF, B_ * T_ / 16, FF / 16, 0);
    }

    const int sw_blocks = (2 * (FF / 16) * 32 + 127) / 128;   // 64 waves

    // ---- 4) s0: belief==0 -> only obs half of W_post1 contributes
    gemm_bf3<<<sw_blocks, 128, 0, stream>>>(
        obs_h, obs_l, T_ * FF, FF, nullptr, nullptr, 0, 0,
        wp1_h + FF, wp1_l + FF, 2 * FF, b_post1,
        nullptr, 0, mid_h, mid_l, FF, 2, FF / 16, 1);
    sample_bf3<<<sw_blocks, 128, 0, stream>>>(
        mid_h, mid_l, wpm_h, wpm_l, b_pm, wps_h, wps_l, b_ps,
        eps0, LAT, out, out_rs);

    // ---- 5) h0
    {
        int n = B_ * FF;
        h0_init<<<(n + 255) / 256, 256, 0, stream>>>(init_h, hA_f, hA_h, hA_l, n);
    }

    // ---- 6) sequential scan (use_post is static: posterior for t<T-1)
    for (int t = 0; t < T_; ++t) {
        const float* s_prev = out + (size_t)t * LAT;            // stride out_rs
        const float* hp_f   = (t & 1) ? hB_f : hA_f;
        bf16 *hp_h = (t & 1) ? hB_h : hA_h, *hp_l = (t & 1) ? hB_l : hA_l;
        float* hn_f = (t & 1) ? hA_f : hB_f;
        bf16 *hn_h = (t & 1) ? hA_h : hB_h, *hn_l = (t & 1) ? hA_l : hB_l;

        // [s_prev | a_t | 0] split, K padded to 544
        {
            int n = B_ * ACAT;
            build_acat<<<(n + 255) / 256, 256, 0, stream>>>(
                s_prev, out_rs, actions + (size_t)t * ACT, T_ * ACT, ac_h, ac_l);
        }
        // gi = relu(acat @ W_gi^T + b_gi)
        gemm_bf3<<<sw_blocks, 128, 0, stream>>>(
            ac_h, ac_l, ACAT, ACAT, nullptr, nullptr, 0, 0,
            wgi_h, wgi_l, ACAT, b_gi,
            nullptr, 0, gi_h, gi_l, FF, 2, FF / 16, 1);
        // GRU cell
        gru_bf3<<<sw_blocks, 128, 0, stream>>>(
            gi_h, gi_l, hp_h, hp_l, hp_f,
            wih_h, wih_l, whh_h, whh_l, bih, bhh,
            hn_f, hn_h, hn_l);

        if (t < T_ - 1) {
            // posterior: mid = relu([h_new | obs_{t+1}] @ W_post1^T + b)
            gemm_bf3<<<sw_blocks, 128, 0, stream>>>(
                hn_h, hn_l, FF, FF,
                obs_h + (size_t)(t + 1) * FF, obs_l + (size_t)(t + 1) * FF, T_ * FF, FF,
                wp1_h, wp1_l, 2 * FF, b_post1,
                nullptr, 0, mid_h, mid_l, FF, 2, FF / 16, 1);
            sample_bf3<<<sw_blocks, 128, 0, stream>>>(
                mid_h, mid_l, wpm_h, wpm_l, b_pm, wps_h, wps_l, b_ps,
                eps_seq + (size_t)t * B_ * LAT, LAT,
                out + (size_t)(t + 1) * LAT, out_rs);
        } else {
            // prior (last step): pmid = relu(h_new @ W_pr1^T + b_pr1)
            gemm_bf3<<<sw_blocks, 128, 0, stream>>>(
                hn_h, hn_l, FF, FF, nullptr, nullptr, 0, 0,
                wpr1_h, wpr1_l, FF, b_pr1,
                nullptr, 0, mid_h, mid_l, FF, 2, FF / 16, 1);
            sample_bf3<<<sw_blocks, 128, 0, stream>>>(
                mid_h, mid_l, wprm_h, wprm_l, b_prm, wprs_h, wprs_l, b_prs,
                eps_seq + (size_t)t * B_ * LAT, LAT,
                out + (size_t)(t + 1) * LAT, out_rs);
        }
    }
}
